// AdditiveAttention_20469814133282
// MI455X (gfx1250) — matrix operand order repaired and checked by reading, not compiled, not run on hardware
//
#include <hip/hip_runtime.h>

// ---------------------------------------------------------------------------
// Additive attention for MI455X (gfx1250, wave32, WMMA).
// Heavy math: two [B*N,256]x[256,256] projections fused into one bf16 WMMA
// accumulator per 16x16 tile (tanh(q+k) is applied to the SUM, so both GEMMs
// share the C fragment). Everything else is HBM-bound streaming.
// ---------------------------------------------------------------------------

typedef __attribute__((ext_vector_type(16))) __bf16 v16bf;
typedef __attribute__((ext_vector_type(8)))  float  v8f;
typedef __attribute__((ext_vector_type(4)))  int    v4i;

union Frag { v16bf v; v4i q[2]; };

__device__ __forceinline__ unsigned short f2bf(float x) {
  // round-to-nearest-even f32 -> bf16
  unsigned u = __float_as_uint(x);
  unsigned r = u + 0x7FFFu + ((u >> 16) & 1u);
  return (unsigned short)(r >> 16);
}

#define NB      32
#define NSEQ    4096
#define DDIM    256
#define HDIM    256
#define BN      (NB * NSEQ)
#define ASTRIDE 264   // 256 + 8 ushorts (16B) pad: 528B row stride staggers LDS banks

// --- Kernel 0: Wq/Wk f32 -> bf16 (done once, reused by all 8192 tiles) -----
__global__ __launch_bounds__(256)
void convert_weights(const float* __restrict__ Wq, const float* __restrict__ Wk,
                     unsigned short* __restrict__ wq, unsigned short* __restrict__ wk) {
  int i = blockIdx.x * 256 + threadIdx.x;   // HDIM*DDIM = 65536 total
  wq[i] = f2bf(Wq[i]);
  wk[i] = f2bf(Wk[i]);
}

// --- Kernel 1: scores[b,n] = sum_h Wv[h] * tanh(q[b,n,h] + k[b,n,h]) -------
// Block = 512 threads = 16 waves; block handles 16 rows of flattened [B*N].
// Wave w handles H columns [16w, 16w+16). 16 WMMAs per wave into one C frag.
__global__ __launch_bounds__(512)
void scores_kernel(const float* __restrict__ Qm, const float* __restrict__ Km,
                   const unsigned short* __restrict__ wq,
                   const unsigned short* __restrict__ wk,
                   const float* __restrict__ Wv,
                   float* __restrict__ scores) {
  __shared__ __align__(16) unsigned short aq[16 * ASTRIDE];
  __shared__ __align__(16) unsigned short ak[16 * ASTRIDE];
  __shared__ float swork[16][16];           // [wave][row] partial scores

  const int    tid     = threadIdx.x;
  const size_t rowBase = (size_t)blockIdx.x * 16;

  // Stage 16x256 q/k tiles into LDS as bf16 (coalesced f32 loads).
  const float* qsrc = Qm + rowBase * DDIM;
  const float* ksrc = Km + rowBase * DDIM;
#pragma unroll
  for (int i = 0; i < 8; ++i) {
    int e = tid + 512 * i;                  // 0..4095
    int r = e >> 8, c = e & 255;
    aq[r * ASTRIDE + c] = f2bf(qsrc[e]);
    ak[r * ASTRIDE + c] = f2bf(ksrc[e]);
  }
  __syncthreads();

  const int wave  = tid >> 5;               // 0..15 (wave32)
  const int lane  = tid & 31;
  const int col   = lane & 15;
  const int hh    = lane >> 4;              // lane half (ISA fragment layouts)
  const int hBase = wave * 16;

  const unsigned short* aqRow = aq + col * ASTRIDE;           // A: row = M = col
  const unsigned short* akRow = ak + col * ASTRIDE;
  const unsigned short* wqRow = wq + (size_t)(hBase + col) * DDIM; // B: col = N
  const unsigned short* wkRow = wk + (size_t)(hBase + col) * DDIM;

  v8f acc = {};                             // shared C frag for q-proj + k-proj
#pragma unroll
  for (int ks = 0; ks < 8; ++ks) {          // K = 256 in steps of 32
    const int k0 = ks * 32;
    Frag AQ, AK, BQ, BK;
    // A 16x32 bf16 layout: VGPR0-3 -> K = 8*half + 0..7 ; VGPR4-7 -> K = 16+8*half + 0..7
    AQ.q[0] = *(const v4i*)(aqRow + k0 + 8 * hh);
    AQ.q[1] = *(const v4i*)(aqRow + k0 + 16 + 8 * hh);
    AK.q[0] = *(const v4i*)(akRow + k0 + 8 * hh);
    AK.q[1] = *(const v4i*)(akRow + k0 + 16 + 8 * hh);
    // B 32x16 bf16 layout (matches A): VGPR0-3 -> K = 8*half + 0..7 ; VGPR4-7 -> K = 16+8*half + 0..7
    BQ.q[0] = *(const v4i*)(wqRow + k0 + 8 * hh);
    BQ.q[1] = *(const v4i*)(wqRow + k0 + 16 + 8 * hh);
    BK.q[0] = *(const v4i*)(wkRow + k0 + 8 * hh);
    BK.q[1] = *(const v4i*)(wkRow + k0 + 16 + 8 * hh);
    acc = __builtin_amdgcn_wmma_f32_16x16x32_bf16(false, AQ.v, false, BQ.v,
                                                  (short)0, acc, false, false);
    acc = __builtin_amdgcn_wmma_f32_16x16x32_bf16(false, AK.v, false, BK.v,
                                                  (short)0, acc, false, false);
  }

  // C layout: lanes 0-15 -> M = v, lanes 16-31 -> M = 8+v; column N = col.
  const float wv = Wv[hBase + col];
  float p[8];
#pragma unroll
  for (int v = 0; v < 8; ++v) p[v] = wv * tanhf(acc[v]);
  // Reduce across the 16 columns (xor of bits 0..3 stays within each half).
#pragma unroll
  for (int m = 1; m < 16; m <<= 1) {
#pragma unroll
    for (int v = 0; v < 8; ++v) p[v] += __shfl_xor(p[v], m, 32);
  }
  if (col == 0) {
#pragma unroll
    for (int v = 0; v < 8; ++v) swork[wave][8 * hh + v] = p[v];
  }
  __syncthreads();
  // Deterministic fixed-order reduce over the 16 H-tiles.
  if (tid < 16) {
    float s = 0.f;
#pragma unroll
    for (int w = 0; w < 16; ++w) s += swork[w][tid];
    scores[rowBase + tid] = s;
  }
}

// --- Kernel 2: per-batch softmax stats + argmax ----------------------------
__global__ __launch_bounds__(256)
void stats_kernel(const float* __restrict__ scores,
                  float* __restrict__ stats, float* __restrict__ out) {
  __shared__ float rmax[256];
  __shared__ int   ridx[256];
  __shared__ float rsum[256];
  const int b = blockIdx.x, t = threadIdx.x;
  const float* s = scores + (size_t)b * NSEQ;

  float m = -__builtin_inff(); int mi = NSEQ;
#pragma unroll 4
  for (int i = 0; i < NSEQ / 256; ++i) {
    int n = t + 256 * i;
    float v = s[n];
    if (v > m) { m = v; mi = n; }           // increasing n -> first index kept
  }
  rmax[t] = m; ridx[t] = mi;
  __syncthreads();
  for (int off = 128; off > 0; off >>= 1) {
    if (t < off) {
      float m2 = rmax[t + off]; int i2 = ridx[t + off];
      if (m2 > rmax[t] || (m2 == rmax[t] && i2 < ridx[t])) { rmax[t] = m2; ridx[t] = i2; }
    }
    __syncthreads();
  }
  const float M = rmax[0];

  float sum = 0.f;
#pragma unroll 4
  for (int i = 0; i < NSEQ / 256; ++i) sum += __expf(s[t + 256 * i] - M);
  rsum[t] = sum;
  __syncthreads();
  for (int off = 128; off > 0; off >>= 1) {
    if (t < off) rsum[t] += rsum[t + off];
    __syncthreads();
  }
  if (t == 0) {
    stats[2 * b]     = M;
    stats[2 * b + 1] = 1.0f / rsum[0];
    out[NB * DDIM + b] = (float)ridx[0];    // aggr_max, appended after out[B,D]
  }
}

// --- Kernel 3: partial weighted value sums (HBM-bound streaming) -----------
__global__ __launch_bounds__(256)
void wsum_partial(const float* __restrict__ scores, const float* __restrict__ values,
                  const float* __restrict__ stats, float* __restrict__ part) {
  __shared__ float alpha[256];
  const int b = blockIdx.y, chunk = blockIdx.x, t = threadIdx.x;
  const int n0 = chunk * 256;
  const float M = stats[2 * b], invZ = stats[2 * b + 1];
  alpha[t] = __expf(scores[(size_t)b * NSEQ + n0 + t] - M) * invZ;
  __syncthreads();
  const float* vb = values + ((size_t)b * NSEQ + n0) * DDIM;
  float acc = 0.f;
#pragma unroll 4
  for (int n = 0; n < 256; ++n) acc = fmaf(alpha[n], vb[(size_t)n * DDIM + t], acc);
  part[((size_t)b * 16 + chunk) * DDIM + t] = acc;
}

// --- Kernel 4: fixed-order reduction of partials -> out[B,D] ---------------
__global__ __launch_bounds__(256)
void wsum_reduce(const float* __restrict__ part, float* __restrict__ out) {
  const int b = blockIdx.x, t = threadIdx.x;
  float s = 0.f;
#pragma unroll
  for (int c = 0; c < 16; ++c) s += part[((size_t)b * 16 + c) * DDIM + t];
  out[b * DDIM + t] = s;
}

extern "C" void kernel_launch(void* const* d_in, const int* in_sizes, int n_in,
                              void* d_out, int out_size, void* d_ws, size_t ws_size,
                              hipStream_t stream) {
  const float* queries = (const float*)d_in[0];   // [B,N,D]
  const float* keys    = (const float*)d_in[1];   // [B,N,D]
  const float* values  = (const float*)d_in[2];   // [B,N,D]
  const float* Wq      = (const float*)d_in[3];   // [H,D]
  const float* Wk      = (const float*)d_in[4];   // [H,D]
  const float* Wv      = (const float*)d_in[5];   // [1,H]
  float* out = (float*)d_out;                     // [B,D] then [B] argmax

  char* ws = (char*)d_ws;
  float*          scores = (float*)ws;          ws += (size_t)BN * sizeof(float);
  unsigned short* wq     = (unsigned short*)ws; ws += (size_t)HDIM * DDIM * sizeof(unsigned short);
  unsigned short* wk     = (unsigned short*)ws; ws += (size_t)HDIM * DDIM * sizeof(unsigned short);
  float*          stats  = (float*)ws;          ws += (size_t)2 * NB * sizeof(float);
  float*          part   = (float*)ws;          // [B][16][D]

  convert_weights<<<(HDIM * DDIM) / 256, 256, 0, stream>>>(Wq, Wk, wq, wk);
  scores_kernel<<<BN / 16, 512, 0, stream>>>(queries, keys, wq, wk, Wv, scores);
  stats_kernel<<<NB, 256, 0, stream>>>(scores, stats, out);
  wsum_partial<<<dim3(16, NB), 256, 0, stream>>>(scores, values, stats, part);
  wsum_reduce<<<NB, 256, 0, stream>>>(part, out);
}
